// S4Layer_41489384079817
// MI455X (gfx1250) — matrix operand order repaired and checked by reading, not compiled, not run on hardware
//
#include <hip/hip_runtime.h>
#include <hip/hip_bf16.h>
#include <stdint.h>

// ---------------------------------------------------------------------------
// Problem constants (from reference)
// ---------------------------------------------------------------------------
#define D_MODEL   1024
#define D_STATE   256
#define NUM_HEADS 2
#define D_HEAD    128
#define BATCH     4
#define SEQ       2048
#define M_TOTAL   (BATCH * SEQ)        // 8192 rows
#define LN_EPS    1e-5f
#define NCHUNK    16                   // scan chunks per sequence
#define CHUNK_LEN (SEQ / NCHUNK)       // 128

typedef __attribute__((ext_vector_type(16))) __bf16 v16bf;
typedef __attribute__((ext_vector_type(8)))  float  v8f;

// gfx1250 async global->LDS copy (ASYNCcnt-tracked), with safe fallback.
// Builtin signature (from clang diagnostic): (v4i AS1*, v4i AS3*, imm off, imm cpol)
#if __has_builtin(__builtin_amdgcn_global_load_async_to_lds_b128)
#define HAVE_ASYNC_LDS 1
typedef int v4i_async __attribute__((__vector_size__(16)));
typedef __attribute__((address_space(1))) v4i_async* g_v4i_ptr;
typedef __attribute__((address_space(3))) v4i_async* l_v4i_ptr;
#else
#define HAVE_ASYNC_LDS 0
#endif

__device__ __forceinline__ void async_copy_b128(const void* g, void* l) {
#if HAVE_ASYNC_LDS
  __builtin_amdgcn_global_load_async_to_lds_b128(
      (g_v4i_ptr)(uintptr_t)g,
      (l_v4i_ptr)(uint32_t)(uintptr_t)l,   // LDS offset lives in low 32 bits
      0, 0);
#else
  *(uint4*)l = *(const uint4*)g;
#endif
}

__device__ __forceinline__ void async_wait0() {
#if HAVE_ASYNC_LDS
#if __has_builtin(__builtin_amdgcn_s_wait_asynccnt)
  __builtin_amdgcn_s_wait_asynccnt(0);
#else
  asm volatile("s_wait_asynccnt 0" ::: "memory");
#endif
#endif
}

// ---------------------------------------------------------------------------
// Prep kernels: weights -> bf16 in [N][K] (pre-transposed) layout so GEMM
// B-tiles are contiguous 16B chunks (async-LDS friendly, no LDS scatter).
// ---------------------------------------------------------------------------

// WpT[n, k] = (bf16) Wp[k, n]     Wp: [1024 x 256]  ->  WpT: [256][1024]
__global__ __launch_bounds__(256) void k_wp_t(const float* __restrict__ Wp,
                                              __bf16* __restrict__ WpT) {
  int id = blockIdx.x * 256 + threadIdx.x;       // 0 .. 262143
  int k = id & (D_MODEL - 1);
  int n = id >> 10;                              // 0..255
  WpT[(size_t)n * D_MODEL + k] = (__bf16)Wp[(size_t)k * D_STATE + n];
}

// WeffT[m, c] = (bf16) sum_e C[h, e, d] * Wo[h*128 + e, m],  c = h*128 + d
// WeffT: [1024][256]  (N-major for GEMM2)
__global__ __launch_bounds__(256) void k_weff(const float* __restrict__ Cp,
                                              const float* __restrict__ Wo,
                                              __bf16* __restrict__ WeffT) {
  int id = blockIdx.x * 256 + threadIdx.x;       // 0 .. 262143
  int c = id & (D_STATE - 1);
  int m = id >> 8;                               // 0..1023
  int hh = c >> 7;
  int d  = c & (D_HEAD - 1);
  const float* crow = Cp + (size_t)hh * D_HEAD * D_HEAD + d;   // stride D_HEAD over e
  const float* wcol = Wo + (size_t)hh * D_HEAD * D_MODEL + m;  // stride D_MODEL over e
  float s = 0.f;
#pragma unroll 8
  for (int e = 0; e < D_HEAD; ++e)
    s += crow[(size_t)e * D_HEAD] * wcol[(size_t)e * D_MODEL];
  WeffT[(size_t)m * D_STATE + c] = (__bf16)s;
}

// ---------------------------------------------------------------------------
// Tiled bf16 WMMA GEMM: Out[M x N] = A(fp32, lda) * B(bf16, [N][K] row major)
// Block = 256 threads (8 wave32), tile 128x128, wave tile 32x64 (2x4 WMMAs).
// MODE 0: Out = (acc + bias[c]) * scl[c]                (input proj -> u)
// MODE 1: Out = acc + bias[c] + res[r*N + c]            (output proj + residual)
// ---------------------------------------------------------------------------
template <int MODE>
__global__ __launch_bounds__(256) void gemm_bf16_wmma(
    const float* __restrict__ A, int lda, int K,
    const __bf16* __restrict__ BmT, int N,
    const float* __restrict__ bias,
    const float* __restrict__ sclOrRes,
    float* __restrict__ Out) {

  __shared__ __align__(16) __bf16 sA[128 * 32];   // [row][k]  8 KB
  __shared__ __align__(16) __bf16 sB[128 * 32];   // [col][k]  8 KB

  const int tid     = threadIdx.x;
  const int rowBase = blockIdx.y * 128;
  const int colBase = blockIdx.x * 128;

  const int lane = tid & 31;
  const int wv   = tid >> 5;        // 0..7
  const int wm   = wv & 3;          // wave M index (4)
  const int wn   = wv >> 2;         // wave N index (2)
  const int mBase = wm * 32;
  const int nBase = wn * 64;
  const int mr = lane & 15;         // row/col within 16x16 tile
  const int hs = lane >> 4;         // half-select (0: lanes 0-15, 1: lanes 16-31)

  v8f acc[2][4] = {};

  // cooperative-load mappings
  const int aCol = (tid & 7) * 4;   // 0..28  (k within tile)
  const int aRow = tid >> 3;        // 0..31

  for (int k0 = 0; k0 < K; k0 += 32) {
    __syncthreads();
    // --- stage B tile: rows n of BmT are contiguous; 512 x 16B chunks,
    //     async global->LDS copies (ASYNCcnt), no VGPR round-trip ---
#pragma unroll
    for (int p = 0; p < 2; ++p) {
      int chunk = tid + p * 256;            // 0..511
      int n  = chunk >> 2;                  // 0..127
      int ke = (chunk & 3) * 8;             // 0,8,16,24 (bf16 elements)
      const __bf16* g = BmT + (size_t)(colBase + n) * (size_t)K + (k0 + ke);
      async_copy_b128(g, &sB[n * 32 + ke]);
    }
    // --- stage A tile (128 x 32) fp32 -> bf16, row major in LDS ---
#pragma unroll
    for (int p = 0; p < 4; ++p) {
      int r = aRow + p * 32;
      const float* gA = A + (size_t)(rowBase + r) * lda + (k0 + aCol);
      const float4 f = *(const float4*)gA;
      if (k0 + 32 < K) __builtin_prefetch(gA + 32, 0, 0);   // next k-tile
      union { __bf16 b[4]; uint2 u2; } pk;
      pk.b[0] = (__bf16)f.x; pk.b[1] = (__bf16)f.y;
      pk.b[2] = (__bf16)f.z; pk.b[3] = (__bf16)f.w;
      *(uint2*)&sA[r * 32 + aCol] = pk.u2;
    }
    async_wait0();
    __syncthreads();

    // --- A fragments: lane row M = mr; lo-lanes K[0:8)+K[16:24), hi-lanes K[8:16)+K[24:32)
    v16bf afr[2];
#pragma unroll
    for (int mt = 0; mt < 2; ++mt) {
      const __bf16* pa = &sA[(mBase + mt * 16 + mr) * 32 + hs * 8];
      union { v16bf v; uint4 q[2]; } u;
      u.q[0] = *(const uint4*)(pa);        // 8 bf16
      u.q[1] = *(const uint4*)(pa + 16);   // +16 elements
      afr[mt] = u.v;
    }
    // --- B fragments + WMMA: lane col N = mr; lo-lanes K[0:8)+K[16:24), hi-lanes K[8:16)+K[24:32)
#pragma unroll
    for (int nt = 0; nt < 4; ++nt) {
      const __bf16* pb = &sB[(nBase + nt * 16 + mr) * 32 + hs * 8];
      union { v16bf v; uint4 q[2]; } u;
      u.q[0] = *(const uint4*)(pb);
      u.q[1] = *(const uint4*)(pb + 16);
      v16bf bfr = u.v;
      acc[0][nt] = __builtin_amdgcn_wmma_f32_16x16x32_bf16(
          false, afr[0], false, bfr, (short)0, acc[0][nt], false, false);
      acc[1][nt] = __builtin_amdgcn_wmma_f32_16x16x32_bf16(
          false, afr[1], false, bfr, (short)0, acc[1][nt], false, false);
    }
  }

  // --- epilogue: C/D layout -> lane<16: N=mr, M=i; lane>=16: N=mr, M=8+i
#pragma unroll
  for (int mt = 0; mt < 2; ++mt) {
#pragma unroll
    for (int nt = 0; nt < 4; ++nt) {
#pragma unroll
      for (int i = 0; i < 8; ++i) {
        int r = rowBase + mBase + mt * 16 + hs * 8 + i;
        int c = colBase + nBase + nt * 16 + mr;
        float v = acc[mt][nt][i];
        if (MODE == 0) {
          Out[(size_t)r * N + c] = (v + bias[c]) * sclOrRes[c];
        } else {
          Out[(size_t)r * N + c] = v + bias[c] + sclOrRes[(size_t)r * N + c];
        }
      }
    }
  }
}

// ---------------------------------------------------------------------------
// Chunked scan (constant per-channel decay lam = sigmoid(lnl[c])):
//   h_t = lam * h_{t-1} + u_t      in-place in the u buffer.
// ---------------------------------------------------------------------------
__device__ __forceinline__ float sigmoidf(float x) { return 1.f / (1.f + expf(-x)); }

// Phase A: local scan per (b, c, chunk); writes local h in place, carryEnd.
__global__ __launch_bounds__(256) void scan_phaseA(float* __restrict__ u,
                                                   const float* __restrict__ lnl,
                                                   float* __restrict__ carryEnd) {
  int g = blockIdx.x * 256 + threadIdx.x;       // 0 .. 16383
  int c = g & (D_STATE - 1);
  int k = (g >> 8) & (NCHUNK - 1);
  int b = g >> 12;
  float lam = sigmoidf(lnl[c]);
  size_t base = ((size_t)(b * SEQ + k * CHUNK_LEN)) * D_STATE + c;
  float h = 0.f;
  for (int t = 0; t < CHUNK_LEN; ++t) {
    size_t idx = base + (size_t)t * D_STATE;
    h = lam * h + u[idx];
    u[idx] = h;
  }
  carryEnd[((size_t)b * D_STATE + c) * NCHUNK + k] = h;
}

// Phase B: sequential scan over chunk carries per channel.
__global__ __launch_bounds__(256) void scan_phaseB(const float* __restrict__ lnl,
                                                   const float* __restrict__ carryEnd,
                                                   float* __restrict__ carryIn) {
  int g = blockIdx.x * 256 + threadIdx.x;       // 0 .. 1023
  int c = g & (D_STATE - 1);
  int b = g >> 8;
  float lam = sigmoidf(lnl[c]);
  float lamT = lam;
#pragma unroll
  for (int i = 0; i < 7; ++i) lamT *= lamT;     // lam^128
  size_t base = ((size_t)b * D_STATE + c) * NCHUNK;
  float cin = 0.f;
  carryIn[base + 0] = 0.f;
  for (int k = 1; k < NCHUNK; ++k) {
    cin = lamT * cin + carryEnd[base + (k - 1)];
    carryIn[base + k] = cin;
  }
}

// Phase C: fixup  h_t += lam^(t_local+1) * carry_in.
__global__ __launch_bounds__(256) void scan_phaseC(float* __restrict__ u,
                                                   const float* __restrict__ lnl,
                                                   const float* __restrict__ carryIn) {
  int g = blockIdx.x * 256 + threadIdx.x;
  int c = g & (D_STATE - 1);
  int k = (g >> 8) & (NCHUNK - 1);
  int b = g >> 12;
  if (k == 0) return;
  float cin = carryIn[((size_t)b * D_STATE + c) * NCHUNK + k];
  float lam = sigmoidf(lnl[c]);
  size_t base = ((size_t)(b * SEQ + k * CHUNK_LEN)) * D_STATE + c;
  float p = lam;
  for (int t = 0; t < CHUNK_LEN; ++t) {
    size_t idx = base + (size_t)t * D_STATE;
    u[idx] += p * cin;
    p *= lam;
  }
}

// ---------------------------------------------------------------------------
// LayerNorm over last dim (1024), in place on d_out. One 256-thread block/row.
// ---------------------------------------------------------------------------
__global__ __launch_bounds__(256) void ln_kernel(float* __restrict__ z,
                                                 const float* __restrict__ gamma,
                                                 const float* __restrict__ beta) {
  __shared__ float red[256];
  __shared__ float red2[256];
  __shared__ float sMean, sRstd;
  const int row = blockIdx.x;
  const int tid = threadIdx.x;
  float* zr = z + (size_t)row * D_MODEL;

  float v[4], s = 0.f, s2 = 0.f;
#pragma unroll
  for (int j = 0; j < 4; ++j) {
    v[j] = zr[tid + j * 256];
    s += v[j];
    s2 += v[j] * v[j];
  }
  red[tid] = s;
  red2[tid] = s2;
  __syncthreads();
  for (int off = 128; off > 0; off >>= 1) {
    if (tid < off) { red[tid] += red[tid + off]; red2[tid] += red2[tid + off]; }
    __syncthreads();
  }
  if (tid == 0) {
    float mean = red[0] * (1.0f / D_MODEL);
    float var  = red2[0] * (1.0f / D_MODEL) - mean * mean;
    sMean = mean;
    sRstd = rsqrtf(var + LN_EPS);
  }
  __syncthreads();
  const float mean = sMean, rstd = sRstd;
#pragma unroll
  for (int j = 0; j < 4; ++j) {
    int m = tid + j * 256;
    zr[m] = (v[j] - mean) * rstd * gamma[m] + beta[m];
  }
}

// ---------------------------------------------------------------------------
// Host-side launch
// ---------------------------------------------------------------------------
extern "C" void kernel_launch(void* const* d_in, const int* in_sizes, int n_in,
                              void* d_out, int out_size, void* d_ws, size_t ws_size,
                              hipStream_t stream) {
  (void)in_sizes; (void)n_in; (void)out_size; (void)ws_size;

  const float* x     = (const float*)d_in[0];   // [4,2048,1024]
  const float* Wp    = (const float*)d_in[1];   // [1024,256]
  const float* bp    = (const float*)d_in[2];   // [256]
  const float* Wo    = (const float*)d_in[3];   // [256,1024]
  const float* bo    = (const float*)d_in[4];   // [1024]
  const float* lnl   = (const float*)d_in[5];   // [2,128]
  const float* Bparm = (const float*)d_in[6];   // [2,128]
  const float* Cparm = (const float*)d_in[7];   // [2,128,128]
  const float* gamma = (const float*)d_in[8];   // [1024]
  const float* beta  = (const float*)d_in[9];   // [1024]
  float* out = (float*)d_out;

  // workspace layout
  char* ws = (char*)d_ws;
  float*  u        = (float*)(ws);                                  // 8 MB (u, then h in place)
  __bf16* WpT      = (__bf16*)(ws + (size_t)8  * 1024 * 1024);      // 512 KB [256][1024]
  __bf16* WeffT    = (__bf16*)(ws + (size_t)9  * 1024 * 1024);      // 512 KB [1024][256]
  float*  carryEnd = (float*)(ws + (size_t)10 * 1024 * 1024);       // 64 KB
  float*  carryIn  = (float*)(ws + (size_t)10 * 1024 * 1024 + 64 * 1024);

  // 1) weight preparation (pre-transposed bf16 [N][K] layouts)
  k_wp_t<<<(D_MODEL * D_STATE) / 256, 256, 0, stream>>>(Wp, WpT);
  k_weff<<<(D_STATE * D_MODEL) / 256, 256, 0, stream>>>(Cparm, Wo, WeffT);

  // 2) input projection: u = (x @ Wp + bp) * B_param   [8192 x 256], K=1024
  gemm_bf16_wmma<0><<<dim3(D_STATE / 128, M_TOTAL / 128), 256, 0, stream>>>(
      x, D_MODEL, D_MODEL, WpT, D_STATE, bp, Bparm, u);

  // 3) chunked diagonal scan (in place in u)
  scan_phaseA<<<(BATCH * D_STATE * NCHUNK) / 256, 256, 0, stream>>>(u, lnl, carryEnd);
  scan_phaseB<<<(BATCH * D_STATE) / 256, 256, 0, stream>>>(lnl, carryEnd, carryIn);
  scan_phaseC<<<(BATCH * D_STATE * NCHUNK) / 256, 256, 0, stream>>>(u, lnl, carryIn);

  // 4) output projection + residual: z = h @ Weff + bo + x   [8192 x 1024], K=256
  gemm_bf16_wmma<1><<<dim3(D_MODEL / 128, M_TOTAL / 128), 256, 0, stream>>>(
      u, D_STATE, D_STATE, WeffT, D_MODEL, bo, x, out);

  // 5) LayerNorm in place on d_out
  ln_kernel<<<M_TOTAL, 256, 0, stream>>>(out, gamma, beta);
}
